// MultiHeadAttention_26353919328576
// MI455X (gfx1250) — compile-verified, matrix operand order checked by reading, not run on hardware
//
#include <hip/hip_runtime.h>

// ---------------------------------------------------------------------------
// Multi-head attention for MI455X (gfx1250, wave32, WMMA 16x16x32 f16,
// async global->LDS staging, double-buffered).
//
//   1. convert_weights : Wq/Wk/Wv ([H,DM,D] f32), Wo ([DM,DM] f32) -> f16,
//                        transposed to [N][K] (WMMA B-frag K-pairs contiguous)
//   2. gemm_kernel x3  : Qh = Q*Wq + bq -> f16 [B,H,S,D]
//                        Kh = K*Wk + bk -> f16 [B,H,S,D]
//                        Vt = V*Wv + bv -> f16 [B,H,D,S]  (transposed)
//      B-tile (64x32 f16, shared by all 4 waves) staged via
//      global_load_async_to_lds_b128, double buffered (s_wait_asynccnt).
//   3. attn_kernel     : flash attention, 4 waves x 16 query rows per block,
//                        K/V 32-key tiles async-staged to LDS + shared by the
//                        4 waves; online softmax; P transposed C->A via LDS.
//   4. gemm_kernel     : out = X*Wo + bo -> f32 (d_out)
//
// Workspace (f16 elems): WqT,WkT,WvT,WoT @4x1M, Qh,Kh,Vt,X @4x8M = 72 MB.
// ---------------------------------------------------------------------------

typedef __attribute__((ext_vector_type(16))) _Float16 v16h;
typedef __attribute__((ext_vector_type(8)))  float    v8f;

#define WMMA_F16(a, b, c) \
  __builtin_amdgcn_wmma_f32_16x16x32_f16(false, (a), false, (b), (short)0, (c), false, false)

// K index base for 16-bit A/B fragments (ISA 7.12.2): slot v of lane-half hi
// holds pair (k, k+1); pairs form two contiguous 16B chunks per fragment.
__device__ __forceinline__ int frag_kbase(int v, int hi) {
  return 2 * v + ((v & 4) ? 8 : 0) + (hi ? 8 : 0);
}

template <typename T>
__device__ __forceinline__ v16h load_frag_rm(const T* __restrict__ base,
                                             int ld, int row, int k0, int hi) {
  v16h f;
  const T* p = base + (size_t)row * (size_t)ld + k0;
#pragma unroll
  for (int v = 0; v < 8; ++v) {
    int kb = frag_kbase(v, hi);
    f[2 * v]     = (_Float16)p[kb];
    f[2 * v + 1] = (_Float16)p[kb + 1];
  }
  return f;
}

// Async copy of one 16-byte chunk global -> LDS (ASYNCcnt-tracked).
// LDS destination offset = low 32 bits of the generic shared pointer
__device__ __forceinline__ void async_copy_b128(const void* gsrc, void* lds_dst) {
  unsigned l = (unsigned)(uintptr_t)lds_dst;
  asm volatile("global_load_async_to_lds_b128 %0, %1, off"
               :: "v"(l), "v"(gsrc) : "memory");
}

// ---------------------------------------------------------------------------
// 1. Weight conversion f32 -> f16, transposed to [N][K] (K = 1024).
//    mode 0: W [H][K][D], n = h*64+d      mode 1: W [K][N]
// ---------------------------------------------------------------------------
__global__ void convert_weights(const float* __restrict__ W,
                                _Float16* __restrict__ WT, int mode) {
  int idx = blockIdx.x * 256 + threadIdx.x;
  int n = idx >> 10, k = idx & 1023;
  float v;
  if (mode == 0) {
    int h = n >> 6, d = n & 63;
    v = W[((size_t)h * 1024 + k) * 64 + d];
  } else {
    v = W[(size_t)k * 1024 + n];
  }
  WT[idx] = (_Float16)v;
}

// ---------------------------------------------------------------------------
// 2/4. GEMM: C[8192,1024] = A * BT^T + bias.  Block = 128 threads = 4 waves,
// each wave a 16x64 C tile.  B k-tile (64 cols x 32 k = 4 KB) is identical
// for all 4 waves -> async-staged to LDS, double buffered.
// MODE 0: f16 [B,H,S,D]   MODE 1: f16 [B,H,D,S] (packed b128 stores)
// MODE 2: f32 [M,N]
// ---------------------------------------------------------------------------
template <typename AT, int MODE>
__global__ void gemm_kernel(const AT* __restrict__ A,
                            const _Float16* __restrict__ BT,
                            const float* __restrict__ bias,
                            void* __restrict__ Out) {
  __shared__ alignas(16) _Float16 ldsB[2][64 * 32];  // 2 x 4 KB
  __shared__ alignas(16) char stage[4][4096];        // epilogue staging

  const int tid = threadIdx.x;
  const int lane = tid & 31, wave = tid >> 5;
  const int hi = lane >> 4, ln = lane & 15;
  const int row0 = blockIdx.x * 64 + wave * 16;
  const int col0 = blockIdx.y * 64;
  const int arow = row0 + ln;

  // stage B tile for k-step `k0` into bbuf: 256 x 16B chunks, 2 per thread
  auto stageB = [&](int k0, _Float16* bbuf) {
#pragma unroll
    for (int i = 0; i < 2; ++i) {
      int c = tid + i * 128;
      int rn = c >> 2, part = (c & 3) * 8;
      async_copy_b128(BT + (size_t)(col0 + rn) * 1024 + k0 + part,
                      bbuf + rn * 32 + part);
    }
  };

  v8f c[4] = {v8f{}, v8f{}, v8f{}, v8f{}};

  stageB(0, ldsB[0]);
  for (int kt = 0; kt < 32; ++kt) {
    const int buf = kt & 1;
    if (kt + 1 < 32) {
      stageB((kt + 1) * 32, ldsB[buf ^ 1]);
      asm volatile("s_wait_asynccnt 0x2" ::: "memory");   // tile kt landed
    } else {
      asm volatile("s_wait_asynccnt 0x0" ::: "memory");
    }
    __syncthreads();                                      // visible block-wide

    v16h a = load_frag_rm(A, 1024, arow, kt * 32, hi);
#pragma unroll
    for (int g = 0; g < 4; ++g) {
      v16h b = load_frag_rm(&ldsB[buf][0], 32, g * 16 + ln, 0, hi);
      c[g] = WMMA_F16(a, b, c[g]);
    }
    __syncthreads();                                      // WAR on ldsB
  }

  // ---- epilogue -----------------------------------------------------------
  if (MODE == 1) {
    // Vt [B,H,D,S]: fixed lane -> fixed (h,d); rows r -> 8 consecutive s.
#pragma unroll
    for (int g = 0; g < 4; ++g) {
      int n = col0 + g * 16 + ln, h = n >> 6, d = n & 63;
      int m0 = row0 + hi * 8, b = m0 >> 10, s0 = m0 & 1023;
      alignas(16) _Float16 pk[8];
#pragma unroll
      for (int r = 0; r < 8; ++r) pk[r] = (_Float16)(c[g][r] + bias[n]);
      _Float16* dst = (_Float16*)Out + (((size_t)b * 16 + h) * 64 + d) * 1024 + s0;
      *(uint4*)dst = *(const uint4*)pk;
    }
  } else if (MODE == 0) {
    // Qh/Kh [B,H,S,D]: stage 16x64 f16 tile in LDS, then b128 stores.
    _Float16* st = (_Float16*)stage[wave];
#pragma unroll
    for (int g = 0; g < 4; ++g)
#pragma unroll
      for (int r = 0; r < 8; ++r) {
        int n = col0 + g * 16 + ln;
        st[(r + hi * 8) * 64 + g * 16 + ln] = (_Float16)(c[g][r] + bias[n]);
      }
    asm volatile("s_wait_dscnt 0x0" ::: "memory");
    int h0 = col0 >> 6;
#pragma unroll
    for (int i = 0; i < 4; ++i) {          // 128 chunks of 16B, 4 per lane
      int ch = lane + i * 32;
      int row = ch >> 3, off8 = (ch & 7) * 8;
      int m = row0 + row, b = m >> 10, s = m & 1023;
      _Float16* dst = (_Float16*)Out + (((size_t)b * 16 + h0) * 1024 + s) * 64 + off8;
      *(uint4*)dst = *(const uint4*)(st + row * 64 + off8);
    }
  } else {
    // f32 [M,N]: stage 16x64 f32 tile, then b128 stores.
    float* st = (float*)stage[wave];
#pragma unroll
    for (int g = 0; g < 4; ++g)
#pragma unroll
      for (int r = 0; r < 8; ++r) {
        int n = col0 + g * 16 + ln;
        st[(r + hi * 8) * 64 + g * 16 + ln] = c[g][r] + bias[n];
      }
    asm volatile("s_wait_dscnt 0x0" ::: "memory");
#pragma unroll
    for (int i = 0; i < 8; ++i) {          // 256 chunks of 16B, 8 per lane
      int ch = lane + i * 32;
      int row = ch >> 4, off4 = (ch & 15) * 4;
      float* dst = (float*)Out + (size_t)(row0 + row) * 1024 + col0 + off4;
      *(uint4*)dst = *(const uint4*)(st + row * 64 + off4);
    }
  }
}

// ---------------------------------------------------------------------------
// 3. Flash attention. Block = 128 threads (4 waves) covers 64 query rows of
// one (b,h). 32-key K/V tiles async-staged to LDS (shared by all 4 waves),
// double buffered; online softmax; causal tile-count is block-uniform
// (per-wave out-of-range tiles are fully masked -> contribute 0).
// ---------------------------------------------------------------------------
__global__ void attn_kernel(const _Float16* __restrict__ Qh,
                            const _Float16* __restrict__ Kh,
                            const _Float16* __restrict__ Vt,
                            const int* __restrict__ pad,
                            _Float16* __restrict__ X) {
  const int S = 1024, D = 64, H = 16;
  __shared__ alignas(16) _Float16 ldsK[2][32 * 64];   // 2 x 4 KB  [key][d]
  __shared__ alignas(16) _Float16 ldsV[2][64 * 32];   // 2 x 4 KB  [d][key]
  __shared__ alignas(16) _Float16 ldsP[4][16 * 32];   // per-wave P tile

  const int tid = threadIdx.x;
  const int lane = tid & 31, wave = tid >> 5;
  const int hi = lane >> 4, ln = lane & 15;
  const int h = blockIdx.y, b = blockIdx.z;
  const int qbase = blockIdx.x * 64 + wave * 16;

  const _Float16* Qp = Qh + ((size_t)b * H + h) * S * D;
  const _Float16* Kp = Kh + ((size_t)b * H + h) * S * D;
  const _Float16* Vp = Vt + ((size_t)b * H + h) * D * S;   // [D][S]

  const int p = pad[b];
  const int kvalid = (p == 0) ? 0 : (S - p);   // pad==0 masks ALL keys (quirk)

  // stage one 32-key K/V tile: 4 async b128 per thread (2 K + 2 V)
  auto stageKV = [&](int tbase, _Float16* kbuf, _Float16* vbuf) {
    const _Float16* sk = Kp + (size_t)tbase * D + tid * 8;   // K: contiguous 4KB
    async_copy_b128(sk, kbuf + tid * 8);
    async_copy_b128(sk + 1024, kbuf + tid * 8 + 1024);
#pragma unroll
    for (int i = 0; i < 2; ++i) {                            // V: 64B rows
      int ch = tid + i * 128;
      int d = ch >> 2, part = (ch & 3) * 8;
      async_copy_b128(Vp + (size_t)d * S + tbase + part, vbuf + d * 32 + part);
    }
  };

  // Q fragments for this wave's 16 rows (K = 64 -> two A frags)
  v16h qa0 = load_frag_rm(Qp, D, qbase + ln, 0, hi);
  v16h qa1 = load_frag_rm(Qp, D, qbase + ln, 32, hi);

  float mrun[8], lrun[8];
  v8f acc[4] = {v8f{}, v8f{}, v8f{}, v8f{}};
#pragma unroll
  for (int r = 0; r < 8; ++r) { mrun[r] = -1e30f; lrun[r] = 0.0f; }

  const float scale = 0.03125f;                 // 1/sqrt(S)
  const int nt = (blockIdx.x * 64 + 95) / 32;   // block-uniform causal bound

  stageKV(0, ldsK[0], ldsV[0]);
  for (int kt = 0; kt < nt; ++kt) {
    const int buf = kt & 1;
    const int tbase = kt * 32;
    if (kt + 1 < nt) {
      stageKV((kt + 1) * 32, ldsK[buf ^ 1], ldsV[buf ^ 1]);
      asm volatile("s_wait_asynccnt 0x4" ::: "memory");   // tile kt landed
    } else {
      asm volatile("s_wait_asynccnt 0x0" ::: "memory");
    }
    __syncthreads();

    // ---- scores: 16x32 = two C frags, K = 64 (2 WMMA steps each)
    const _Float16* kb = &ldsK[buf][0];
    v16h kb00 = load_frag_rm(kb, 64, ln,      0,  hi);
    v16h kb01 = load_frag_rm(kb, 64, ln,      32, hi);
    v16h kb10 = load_frag_rm(kb, 64, 16 + ln, 0,  hi);
    v16h kb11 = load_frag_rm(kb, 64, 16 + ln, 32, hi);
    v8f s0 = {}, s1 = {};
    s0 = WMMA_F16(qa0, kb00, s0);
    s0 = WMMA_F16(qa1, kb01, s0);
    s1 = WMMA_F16(qa0, kb10, s1);
    s1 = WMMA_F16(qa1, kb11, s1);

    // ---- scale + causal/pad mask
    int key0 = tbase + ln, key1 = key0 + 16;
#pragma unroll
    for (int r = 0; r < 8; ++r) {
      int m = qbase + r + hi * 8;
      float v0 = s0[r] * scale, v1 = s1[r] * scale;
      if (key0 > m || key0 >= kvalid) v0 -= 1e9f;
      if (key1 > m || key1 >= kvalid) v1 -= 1e9f;
      s0[r] = v0; s1[r] = v1;
    }

    // ---- online softmax (row = one 16-lane half of the wave)
#pragma unroll
    for (int r = 0; r < 8; ++r) {
      float t = fmaxf(s0[r], s1[r]);
#pragma unroll
      for (int off = 8; off >= 1; off >>= 1) t = fmaxf(t, __shfl_xor(t, off, 32));
      float mnew = fmaxf(mrun[r], t);
      float p0 = __expf(s0[r] - mnew);
      float p1 = __expf(s1[r] - mnew);
      float ssum = p0 + p1;
#pragma unroll
      for (int off = 8; off >= 1; off >>= 1) ssum += __shfl_xor(ssum, off, 32);
      float cs = __expf(mrun[r] - mnew);
      lrun[r] = lrun[r] * cs + ssum;
      mrun[r] = mnew;
#pragma unroll
      for (int g = 0; g < 4; ++g) acc[g][r] *= cs;
      s0[r] = p0; s1[r] = p1;
    }

    // ---- P: C-layout -> A-layout via per-wave LDS region
    _Float16* pw = &ldsP[wave][0];
#pragma unroll
    for (int r = 0; r < 8; ++r) {
      int m = r + hi * 8;
      pw[m * 32 + ln]      = (_Float16)s0[r];
      pw[m * 32 + 16 + ln] = (_Float16)s1[r];
    }
    asm volatile("s_wait_dscnt 0x0" ::: "memory");   // LDS in-order per wave
    v16h pa = load_frag_rm(pw, 32, ln, 0, hi);

    // ---- O += P * V   (V B-frags from LDS [d][key], K = 32)
    const _Float16* vb = &ldsV[buf][0];
#pragma unroll
    for (int g = 0; g < 4; ++g) {
      v16h vf = load_frag_rm(vb, 32, g * 16 + ln, 0, hi);
      acc[g] = WMMA_F16(pa, vf, acc[g]);
    }
    __syncthreads();                                 // WAR on ldsK/ldsV
  }

  // ---- epilogue: normalize, stage 16x64 tile in LDS (reuse ldsK), b128 store
  _Float16* ob = ((_Float16*)ldsK) + wave * 1024;    // 2 KB per wave
#pragma unroll
  for (int g = 0; g < 4; ++g)
#pragma unroll
    for (int r = 0; r < 8; ++r)
      ob[(r + hi * 8) * 64 + g * 16 + ln] = (_Float16)(acc[g][r] / lrun[r]);
  asm volatile("s_wait_dscnt 0x0" ::: "memory");
#pragma unroll
  for (int i = 0; i < 4; ++i) {                      // 128 chunks, 4 per lane
    int ch = lane + i * 32;
    int row = ch >> 3, off8 = (ch & 7) * 8;
    _Float16* dst = X + ((size_t)b * S + qbase - wave * 16 + wave * 16 + row) * 1024
                      + h * 64 + off8;
    dst = X + ((size_t)b * S + (blockIdx.x * 64 + wave * 16) + row) * 1024 + h * 64 + off8;
    *(uint4*)dst = *(const uint4*)(ob + row * 64 + off8);
  }
}

// ---------------------------------------------------------------------------
extern "C" void kernel_launch(void* const* d_in, const int* in_sizes, int n_in,
                              void* d_out, int out_size, void* d_ws, size_t ws_size,
                              hipStream_t stream) {
  const float* Q  = (const float*)d_in[0];
  const float* K  = (const float*)d_in[1];
  const float* V  = (const float*)d_in[2];
  const int*   pad = (const int*)d_in[3];
  const float* Wq = (const float*)d_in[4];
  const float* bq = (const float*)d_in[5];
  const float* Wk = (const float*)d_in[6];
  const float* bk = (const float*)d_in[7];
  const float* Wv = (const float*)d_in[8];
  const float* bv = (const float*)d_in[9];
  const float* Wo = (const float*)d_in[10];
  const float* bo = (const float*)d_in[11];

  const size_t MEG = 1u << 20;
  _Float16* ws  = (_Float16*)d_ws;
  _Float16* WqT = ws + 0 * MEG;
  _Float16* WkT = ws + 1 * MEG;
  _Float16* WvT = ws + 2 * MEG;
  _Float16* WoT = ws + 3 * MEG;
  _Float16* Qh  = ws + 4 * MEG;    // [B,H,S,D]
  _Float16* Kh  = ws + 12 * MEG;   // [B,H,S,D]
  _Float16* Vt  = ws + 20 * MEG;   // [B,H,D,S]
  _Float16* X   = ws + 28 * MEG;   // [B,S,DM]

  convert_weights<<<4096, 256, 0, stream>>>(Wq, WqT, 0);
  convert_weights<<<4096, 256, 0, stream>>>(Wk, WkT, 0);
  convert_weights<<<4096, 256, 0, stream>>>(Wv, WvT, 0);
  convert_weights<<<4096, 256, 0, stream>>>(Wo, WoT, 1);

  dim3 ggrid(8192 / 64, 1024 / 64);
  gemm_kernel<float, 0><<<ggrid, 128, 0, stream>>>(Q, WqT, bq, (void*)Qh);
  gemm_kernel<float, 0><<<ggrid, 128, 0, stream>>>(K, WkT, bk, (void*)Kh);
  gemm_kernel<float, 1><<<ggrid, 128, 0, stream>>>(V, WvT, bv, (void*)Vt);

  attn_kernel<<<dim3(1024 / 64, 16, 8), 128, 0, stream>>>(Qh, Kh, Vt, pad, X);

  gemm_kernel<_Float16, 2><<<ggrid, 128, 0, stream>>>(X, WoT, bo, d_out);
}
